// Teacher_S_64330020159590
// MI455X (gfx1250) — compile-verified, matrix operand order checked by reading, not run on hardware
//
#include <hip/hip_runtime.h>
#include <stdint.h>

#define NNODES 10000
#define NEDGES 320000
#define DIN    512
#define DHID   512
#define DOUTL  256
#define BLK    256

#define MTILES     625                 // ceil(10000/16), all full (10000 = 625*16)
#define MPAD       10048               // 628 tiles -> divisible by 4 tiles/wave
#define WGROUPS    (MPAD / 64)         // 157 wave-groups of 4 M-tiles
#define LDS_STRIDE 40                  // bf16 elems per LDS row (80B, banks spread)
#define TILE_ELEMS (16 * LDS_STRIDE)   // one 16x32 B-tile slot in LDS

typedef __attribute__((ext_vector_type(16))) __bf16 v16bf;
typedef __attribute__((ext_vector_type(8)))  float  v8f;

__device__ __forceinline__ unsigned short f32_to_bf16(float f) {
  union { float f; uint32_t u; } x; x.f = f;
  uint32_t r = x.u + 0x7FFFu + ((x.u >> 16) & 1u);   // round-to-nearest-even
  return (unsigned short)(r >> 16);
}

__global__ void zero_f32(float* __restrict__ p, int n) {
  int i = blockIdx.x * BLK + threadIdx.x;
  if (i < n) p[i] = 0.0f;
}

__global__ void cvt_bf16(const float* __restrict__ s, unsigned short* __restrict__ d, int n) {
  int i = blockIdx.x * BLK + threadIdx.x;
  if (i < n) d[i] = f32_to_bf16(s[i]);
}

// pe[node, f] = w_pe[f, node] + b_pe[f]  (w_pe is [IN, N] row-major)
__global__ void pe_kernel(const float* __restrict__ w_pe, const float* __restrict__ b_pe,
                          float* __restrict__ h0f, unsigned short* __restrict__ h0b) {
  int i = blockIdx.x * BLK + threadIdx.x;
  if (i >= NNODES * DIN) return;
  int node = i / DIN, f = i % DIN;
  float v = w_pe[f * NNODES + node] + b_pe[f];
  h0f[node * DIN + f] = v;
  h0b[node * DIN + f] = f32_to_bf16(v);
}

__global__ void deg_kernel(const int* __restrict__ dst, float* __restrict__ deg) {
  int e = blockIdx.x * BLK + threadIdx.x;
  if (e < NEDGES) atomicAdd(&deg[dst[e]], 1.0f);
}

// edge-parallel segment_sum: one thread per (edge, 4-feature chunk); Din fixed at 512
__global__ void scatter_kernel(const int* __restrict__ src, const int* __restrict__ dst,
                               const float* __restrict__ h, float* __restrict__ agg) {
  int i = blockIdx.x * BLK + threadIdx.x;
  if (i >= NEDGES * (DIN / 4)) return;
  int e = i >> 7;
  int c = (i & 127) << 2;
  float4 v = *(const float4*)(h + src[e] * DIN + c);
  float* a = agg + dst[e] * DIN + c;
  atomicAdd(a + 0, v.x); atomicAdd(a + 1, v.y);
  atomicAdd(a + 2, v.z); atomicAdd(a + 3, v.w);
}

// aggb = bf16( aggf / max(deg,1) )
__global__ void finalize_agg(const float* __restrict__ aggf, const float* __restrict__ deg,
                             unsigned short* __restrict__ aggb) {
  int i = blockIdx.x * BLK + threadIdx.x;
  if (i >= NNODES * DIN) return;
  float dg = deg[i / DIN];
  float s = 1.0f / fmaxf(dg, 1.0f);
  aggb[i] = f32_to_bf16(aggf[i] * s);
}

union Frag { v16bf v; uint4 q[2]; };

// Async-stage the three 16x32 bf16 weight tiles for column-block tileN, K-slice k0,
// into LDS buffer `buf`. 6 lane-wide b128 async copies (3KB), tracked by ASYNCcnt.
__device__ __forceinline__ void stage_tiles(const unsigned short* w0,
                                            const unsigned short* w1,
                                            const unsigned short* w2,
                                            unsigned short* bufBase,
                                            int bRowBase, int Din, int k0, int lane) {
  const unsigned short* w[3] = { w0, w1, w2 };
#pragma unroll
  for (int j = 0; j < 3; ++j) {
#pragma unroll
    for (int i = 0; i < 2; ++i) {
      int chunk = i * 32 + lane;          // 64 x 16B chunks per tile
      int n = chunk >> 2;                 // row within tile (0..15)
      int c = chunk & 3;                  // 16B chunk within 64B row slice
      uint64_t ga = (uint64_t)(uintptr_t)(w[j] + (size_t)(bRowBase + n) * Din + k0 + c * 8);
      uint32_t la = (uint32_t)(uintptr_t)(bufBase + j * TILE_ELEMS + n * LDS_STRIDE + c * 8);
      asm volatile("global_load_async_to_lds_b128 %0, %1, off"
                   :: "v"(la), "v"(ga) : "memory");
    }
  }
}

// Fused SAGE layer GEMM:
//   conv = A@Ws^T + G@Wn^T + bconv ; (leaky) ; out = conv + A@Wr^T + bres
// A,G bf16 [MPAD,Din]; W* bf16 [Dout,Din] row-major. Wave = 4 M-tiles x 1 N-tile,
// B tiles double-buffered in LDS via async global->LDS copies issued by wave 0.
__global__ void __launch_bounds__(BLK)
sage_gemm(const unsigned short* __restrict__ Ab, const unsigned short* __restrict__ Gb,
          const unsigned short* __restrict__ Ws, const unsigned short* __restrict__ Wn,
          const unsigned short* __restrict__ Wr,
          const float* __restrict__ bconv, const float* __restrict__ bres,
          float* __restrict__ outF, unsigned short* __restrict__ outB,
          float* __restrict__ outF2,
          int Din, int Dout, int leaky) {
  __shared__ __align__(16) unsigned short smem[2][3][TILE_ELEMS];

  const int wave  = threadIdx.x >> 5;
  const int lane  = threadIdx.x & 31;
  const int g     = blockIdx.y * 8 + wave;       // wave-group of 4 full M-tiles
  const int tileN = blockIdx.x;
  const int half  = lane >> 4;
  const int mrow  = lane & 15;
  const bool active = g < WGROUPS;               // wave-uniform

  v8f accC[4], accR[4];
#pragma unroll
  for (int t = 0; t < 4; ++t) {
    accC[t] = (v8f){0.f,0.f,0.f,0.f,0.f,0.f,0.f,0.f};
    accR[t] = (v8f){0.f,0.f,0.f,0.f,0.f,0.f,0.f,0.f};
  }

  const int nstages = Din / 32;

  // prologue: stage k-slice 0 into buffer 0
  if (wave == 0) {
    stage_tiles(Ws, Wn, Wr, &smem[0][0][0], tileN * 16, Din, 0, lane);
    asm volatile("s_wait_asynccnt 0x0" ::: "memory");
  }
  __syncthreads();

  int cur = 0;
  for (int s = 0; s < nstages; ++s) {
    // kick off next slice into the other buffer while this one is consumed
    if (wave == 0 && (s + 1) < nstages)
      stage_tiles(Ws, Wn, Wr, &smem[cur ^ 1][0][0], tileN * 16, Din, (s + 1) * 32, lane);

    if (active) {                                 // wave-uniform, EXEC stays all-ones
      const int k0 = s * 32;
      const unsigned short* bbase = &smem[cur][0][mrow * LDS_STRIDE + half * 8];
      Frag fs, fn, fr;
      fs.q[0] = *(const uint4*)(bbase);
      fs.q[1] = *(const uint4*)(bbase + 16);
      fn.q[0] = *(const uint4*)(bbase + TILE_ELEMS);
      fn.q[1] = *(const uint4*)(bbase + TILE_ELEMS + 16);
      fr.q[0] = *(const uint4*)(bbase + 2 * TILE_ELEMS);
      fr.q[1] = *(const uint4*)(bbase + 2 * TILE_ELEMS + 16);
#pragma unroll
      for (int t = 0; t < 4; ++t) {
        const int aOff = ((g * 4 + t) * 16 + mrow) * Din + k0 + half * 8;
        Frag fa, fg;
        fa.q[0] = *(const uint4*)(Ab + aOff); fa.q[1] = *(const uint4*)(Ab + aOff + 16);
        fg.q[0] = *(const uint4*)(Gb + aOff); fg.q[1] = *(const uint4*)(Gb + aOff + 16);
        accC[t] = __builtin_amdgcn_wmma_f32_16x16x32_bf16(false, fa.v, false, fs.v,
                                                          (short)0, accC[t], false, false);
        accC[t] = __builtin_amdgcn_wmma_f32_16x16x32_bf16(false, fg.v, false, fn.v,
                                                          (short)0, accC[t], false, false);
        accR[t] = __builtin_amdgcn_wmma_f32_16x16x32_bf16(false, fa.v, false, fr.v,
                                                          (short)0, accR[t], false, false);
      }
    }

    if (wave == 0) asm volatile("s_wait_asynccnt 0x0" ::: "memory");
    __syncthreads();
    cur ^= 1;
  }

  if (!active) return;

  const int col = tileN * 16 + mrow;
  const float bc = bconv[col];
  const float br = bres[col];
#pragma unroll
  for (int t = 0; t < 4; ++t) {
    int mt = g * 4 + t;
    if (mt >= MTILES) continue;                   // padded tiles: discard (wave-uniform)
#pragma unroll
    for (int i = 0; i < 8; ++i) {
      int row = mt * 16 + half * 8 + i;
      float conv = accC[t][i] + bc;
      if (leaky) conv = conv > 0.0f ? conv : 0.01f * conv;
      float val = conv + accR[t][i] + br;
      int o = row * Dout + col;
      outF[o] = val;
      if (outB)  outB[o]  = f32_to_bf16(val);
      if (outF2) outF2[o] = val;
    }
  }
}

static inline int cdiv(int a, int b) { return (a + b - 1) / b; }

extern "C" void kernel_launch(void* const* d_in, const int* in_sizes, int n_in,
                              void* d_out, int out_size, void* d_ws, size_t ws_size,
                              hipStream_t stream) {
  (void)in_sizes; (void)n_in; (void)out_size; (void)ws_size;

  const int*   ei   = (const int*)d_in[0];
  const int*   srcI = ei;
  const int*   dstI = ei + NEDGES;
  const float* w_pe = (const float*)d_in[1];
  const float* b_pe = (const float*)d_in[2];

  const float* Wf[3][3]; const float* Bc[3]; const float* Br[3];
  int wsz[3];
  for (int l = 0; l < 3; ++l) {
    Wf[l][0] = (const float*)d_in[3 + 5 * l + 0];   // self
    Wf[l][1] = (const float*)d_in[3 + 5 * l + 1];   // neigh
    Bc[l]    = (const float*)d_in[3 + 5 * l + 2];   // b
    Wf[l][2] = (const float*)d_in[3 + 5 * l + 3];   // res
    Br[l]    = (const float*)d_in[3 + 5 * l + 4];   // rb
  }
  wsz[0] = DHID * DIN; wsz[1] = DHID * DHID; wsz[2] = DOUTL * DHID;

  // ---- workspace carve-up (all chunks 16B-multiple); bf16 act buffers padded to MPAD rows ----
  char* p = (char*)d_ws;
  float*          h0f  = (float*)p;          p += (size_t)NNODES * DIN * 4;
  unsigned short* h0b  = (unsigned short*)p; p += (size_t)MPAD * DIN * 2;
  unsigned short* h1b  = (unsigned short*)p; p += (size_t)MPAD * DHID * 2;
  unsigned short* h2b  = (unsigned short*)p; p += (size_t)MPAD * DHID * 2;
  float*          aggf = (float*)p;          p += (size_t)NNODES * DIN * 4;
  unsigned short* aggb = (unsigned short*)p; p += (size_t)MPAD * DIN * 2;
  float*          deg  = (float*)p;          p += (size_t)NNODES * 4;
  unsigned short* Wb[3][3];
  for (int l = 0; l < 3; ++l)
    for (int j = 0; j < 3; ++j) { Wb[l][j] = (unsigned short*)p; p += (size_t)wsz[l] * 2; }

  // ---- d_out regions: (final, mid0, mid1, mid2) ----
  float* outFinal = (float*)d_out;
  float* mid0 = outFinal + (size_t)NNODES * DOUTL;
  float* mid1 = mid0 + (size_t)NNODES * DHID;
  float* mid2 = mid1 + (size_t)NNODES * DHID;

  // weights -> bf16
  for (int l = 0; l < 3; ++l)
    for (int j = 0; j < 3; ++j)
      cvt_bf16<<<cdiv(wsz[l], BLK), BLK, 0, stream>>>(Wf[l][j], Wb[l][j], wsz[l]);

  // positional encoding (h0)
  pe_kernel<<<cdiv(NNODES * DIN, BLK), BLK, 0, stream>>>(w_pe, b_pe, h0f, h0b);

  // degrees (shared by all layers)
  zero_f32<<<cdiv(NNODES, BLK), BLK, 0, stream>>>(deg, NNODES);
  deg_kernel<<<cdiv(NEDGES, BLK), BLK, 0, stream>>>(dstI, deg);

  const int scatterBlocks = cdiv(NEDGES * (DIN / 4), BLK);
  const int gemmY = cdiv(WGROUPS, 8);               // 8 waves/block

  // ---- layer 0: 512 -> 512, leaky ----
  zero_f32<<<cdiv(NNODES * DIN, BLK), BLK, 0, stream>>>(aggf, NNODES * DIN);
  scatter_kernel<<<scatterBlocks, BLK, 0, stream>>>(srcI, dstI, h0f, aggf);
  finalize_agg<<<cdiv(NNODES * DIN, BLK), BLK, 0, stream>>>(aggf, deg, aggb);
  sage_gemm<<<dim3(DHID / 16, gemmY), BLK, 0, stream>>>(
      h0b, aggb, Wb[0][0], Wb[0][1], Wb[0][2], Bc[0], Br[0],
      mid0, h1b, nullptr, DIN, DHID, 1);

  // ---- layer 1: 512 -> 512, leaky ----
  zero_f32<<<cdiv(NNODES * DIN, BLK), BLK, 0, stream>>>(aggf, NNODES * DIN);
  scatter_kernel<<<scatterBlocks, BLK, 0, stream>>>(srcI, dstI, mid0, aggf);
  finalize_agg<<<cdiv(NNODES * DIN, BLK), BLK, 0, stream>>>(aggf, deg, aggb);
  sage_gemm<<<dim3(DHID / 16, gemmY), BLK, 0, stream>>>(
      h1b, aggb, Wb[1][0], Wb[1][1], Wb[1][2], Bc[1], Br[1],
      mid1, h2b, nullptr, DHID, DHID, 1);

  // ---- layer 2: 512 -> 256, no leaky; duplicate into final slot ----
  zero_f32<<<cdiv(NNODES * DIN, BLK), BLK, 0, stream>>>(aggf, NNODES * DIN);
  scatter_kernel<<<scatterBlocks, BLK, 0, stream>>>(srcI, dstI, mid1, aggf);
  finalize_agg<<<cdiv(NNODES * DIN, BLK), BLK, 0, stream>>>(aggf, deg, aggb);
  sage_gemm<<<dim3(DOUTL / 16, gemmY), BLK, 0, stream>>>(
      h2b, aggb, Wb[2][0], Wb[2][1], Wb[2][2], Bc[2], Br[2],
      mid2, nullptr, outFinal, DHID, DOUTL, 0);
}
